// Attention_687194768262
// MI455X (gfx1250) — matrix operand order repaired and checked by reading, not compiled, not run on hardware
//
#include <hip/hip_runtime.h>

// ---------------------------------------------------------------------------
// MHA for MI455X (gfx1250, wave32): all GEMMs via v_wmma_f32_16x16x32_bf16.
// Pipeline: cvt(x) + transpose-cvt(weights) -> QKV GEMM (scatter Q*scale, K,
// V^T as bf16) -> flash attention (online softmax, bias folded into WMMA C
// init) -> output projection GEMM (f32 out).
// GEMM waves use 32x64 tiles (8 WMMAs / k-step, ~22 FLOP/byte from L2).
// ---------------------------------------------------------------------------

typedef __attribute__((ext_vector_type(16))) __bf16 v16bf;
typedef __attribute__((ext_vector_type(8)))  __bf16 v8bf;
typedef __attribute__((ext_vector_type(8)))  float  v8f;
typedef __attribute__((ext_vector_type(4)))  float  v4f;

static constexpr int B_     = 8;
static constexpr int N_     = 1024;
static constexpr int DIM_   = 512;
static constexpr int HEADS_ = 8;
static constexpr int DH_    = 64;
static constexpr int HID_   = 512;
static constexpr int M_     = B_ * N_;      // 8192 rows
static constexpr int QKV3_  = HID_ * 3;     // 1536 cols
#define SCALE_F 0.125f                      // 64^-0.5

__device__ __forceinline__ __bf16 f2bf(float f) {
  unsigned u = __builtin_bit_cast(unsigned, f);
  u += 0x7FFFu + ((u >> 16) & 1u);          // round-to-nearest-even
  unsigned short h = (unsigned short)(u >> 16);
  return __builtin_bit_cast(__bf16, h);
}

// Build a 16-element bf16 fragment from two 16-byte chunks.
__device__ __forceinline__ v16bf frag_from(const __bf16* p0, const __bf16* p1) {
  v8bf lo = *(const v8bf*)p0;
  v8bf hi = *(const v8bf*)p1;
  v16bf r;
#pragma unroll
  for (int i = 0; i < 8; ++i) { r[i] = lo[i]; r[i + 8] = hi[i]; }
  return r;
}

__device__ __forceinline__ v8f wmma_bf16(v16bf a, v16bf b, v8f c) {
  return __builtin_amdgcn_wmma_f32_16x16x32_bf16(false, a, false, b,
                                                 (short)0, c, false, false);
}

// ---------------------------------------------------------------- prep ----
__global__ __launch_bounds__(256) void k_cvt_bf16(const float* __restrict__ in,
                                                  __bf16* __restrict__ out, int n) {
  int i = (blockIdx.x * 256 + threadIdx.x) * 4;
  if (i + 3 < n) {
    v4f v = *(const v4f*)(in + i);
    out[i + 0] = f2bf(v[0]); out[i + 1] = f2bf(v[1]);
    out[i + 2] = f2bf(v[2]); out[i + 3] = f2bf(v[3]);
  }
}

// out[n*K + k] = in[k*Ncols + n]  (B^T in bf16 so B-fragments are contiguous)
__global__ __launch_bounds__(256) void k_transpose_cvt(const float* __restrict__ in,
                                                       __bf16* __restrict__ out,
                                                       int K, int Ncols) {
  int n = blockIdx.x;
  for (int k = threadIdx.x; k < K; k += 256)
    out[(size_t)n * K + k] = f2bf(in[(size_t)k * Ncols + n]);
}

// ------------------------------------------------------------- QKV GEMM ----
// C[8192,1536] = xb[8192,512] * wT[1536,512]^T ; wave tile 32x64; scatter epi.
__global__ __launch_bounds__(256) void k_qkv_gemm(const __bf16* __restrict__ xb,
                                                  const __bf16* __restrict__ wT,
                                                  __bf16* __restrict__ Qs,
                                                  __bf16* __restrict__ Kb,
                                                  __bf16* __restrict__ Vt) {
  const int lane = threadIdx.x & 31;
  const int wid  = threadIdx.x >> 5;
  const int gw   = blockIdx.x * 8 + wid;
  const int TN   = QKV3_ / 64;              // 24 col tiles
  const int tm = gw / TN, tn = gw % TN;
  const int m0 = tm * 32, n0 = tn * 64;
  const int lrow = lane & 15;
  const int half = lane >> 4;
  const int kbA  = half << 3;               // A: +0 / +8
  const int kbB  = half << 3;               // B: +0 / +8

  v8f acc[2][4] = {};
#pragma unroll 2
  for (int k0 = 0; k0 < DIM_; k0 += 32) {
    const __bf16* a0p = xb + (size_t)(m0 + lrow) * DIM_ + k0 + kbA;
    const __bf16* a1p = xb + (size_t)(m0 + 16 + lrow) * DIM_ + k0 + kbA;
    v16bf a0 = frag_from(a0p, a0p + 16);
    v16bf a1 = frag_from(a1p, a1p + 16);
    v16bf bf[4];
#pragma unroll
    for (int j = 0; j < 4; ++j) {
      const __bf16* bp = wT + (size_t)(n0 + j * 16 + lrow) * DIM_ + k0 + kbB;
      bf[j] = frag_from(bp, bp + 16);
    }
#pragma unroll
    for (int j = 0; j < 4; ++j) {
      acc[0][j] = wmma_bf16(a0, bf[j], acc[0][j]);
      acc[1][j] = wmma_bf16(a1, bf[j], acc[1][j]);
    }
  }
#pragma unroll
  for (int i = 0; i < 2; ++i)
#pragma unroll
    for (int j = 0; j < 4; ++j)
#pragma unroll
      for (int r = 0; r < 8; ++r) {
        int row = m0 + i * 16 + r + (half << 3);
        int col = n0 + j * 16 + lrow;
        float v = acc[i][j][r];
        int seg = col >> 9, rem = col & 511;
        int h = rem >> 6, d = rem & 63;
        int b = row >> 10, nn = row & 1023;
        int bh = b * HEADS_ + h;
        if (seg == 0)      Qs[((size_t)bh * N_ + nn) * DH_ + d] = f2bf(v * SCALE_F);
        else if (seg == 1) Kb[((size_t)bh * N_ + nn) * DH_ + d] = f2bf(v);
        else               Vt[((size_t)bh * DH_ + d) * N_ + nn] = f2bf(v);
      }
}

// ------------------------------------------------------ flash attention ----
// One wave = 16 query rows of one (b,h). j-steps of 32 keys.
__global__ __launch_bounds__(256) void k_flash(const __bf16* __restrict__ Qs,
                                               const __bf16* __restrict__ Kb,
                                               const __bf16* __restrict__ Vt,
                                               const float*  __restrict__ bias,
                                               __bf16* __restrict__ Ob) {
  __shared__ __align__(16) __bf16 psh[8][16 * 32];  // per-wave P transpose buf
  const int lane = threadIdx.x & 31;
  const int wid  = threadIdx.x >> 5;
  const int gw   = blockIdx.x * 8 + wid;
  const int bh = gw >> 6;                   // 0..63
  const int i0 = (gw & 63) * 16;
  const int b  = bh >> 3, h = bh & 7;
  const int lrow = lane & 15;
  const int half = lane >> 4;

  const __bf16* Qbh = Qs + (size_t)bh * N_ * DH_;
  const __bf16* Kbh = Kb + (size_t)bh * N_ * DH_;
  const __bf16* Vbh = Vt + (size_t)bh * DH_ * N_;
  const float*  Bh  = bias + (size_t)h * N_ * N_;

  v16bf qa[2];                              // Q 16x64 -> two 16x32 A-frags
#pragma unroll
  for (int f = 0; f < 2; ++f) {
    const __bf16* p = Qbh + (size_t)(i0 + lrow) * DH_ + f * 32 + (half << 3);
    qa[f] = frag_from(p, p + 16);
  }

  float m[8], l[8];
#pragma unroll
  for (int r = 0; r < 8; ++r) { m[r] = -3.0e38f; l[r] = 0.f; }
  v8f oacc[4] = {};
  __bf16* pbase = &psh[wid][0];
  const float LOG2E = 1.4426950408889634f;

  for (int j0 = 0; j0 < N_; j0 += 32) {
    v8f s[2];                               // bias folded into C init
#pragma unroll
    for (int t = 0; t < 2; ++t)
#pragma unroll
      for (int r = 0; r < 8; ++r)
        s[t][r] = Bh[(size_t)(i0 + r + (half << 3)) * N_ + (j0 + t * 16 + lrow)];

#pragma unroll
    for (int t = 0; t < 2; ++t) {           // S = Q * K^T (+bias)
      const __bf16* kp0 = Kbh + (size_t)(j0 + t * 16 + lrow) * DH_ + (half << 3);
      v16bf kf0 = frag_from(kp0, kp0 + 16);
      v16bf kf1 = frag_from(kp0 + 32, kp0 + 48);
      s[t] = wmma_bf16(qa[0], kf0, s[t]);
      s[t] = wmma_bf16(qa[1], kf1, s[t]);
    }

    float scale_r[8];                       // online softmax (rows in 16-lane halves)
#pragma unroll
    for (int r = 0; r < 8; ++r) {
      float rm = fmaxf(s[0][r], s[1][r]);
#pragma unroll
      for (int off = 8; off >= 1; off >>= 1)
        rm = fmaxf(rm, __shfl_xor(rm, off, 32));
      float mn = fmaxf(m[r], rm);
      scale_r[r] = __builtin_exp2f((m[r] - mn) * LOG2E);
      m[r] = mn;
    }
    float p0[8], p1[8];
#pragma unroll
    for (int r = 0; r < 8; ++r) {
      p0[r] = __builtin_exp2f((s[0][r] - m[r]) * LOG2E);
      p1[r] = __builtin_exp2f((s[1][r] - m[r]) * LOG2E);
      float rs = p0[r] + p1[r];
#pragma unroll
      for (int off = 8; off >= 1; off >>= 1)
        rs += __shfl_xor(rs, off, 32);
      l[r] = l[r] * scale_r[r] + rs;
    }
#pragma unroll
    for (int t2 = 0; t2 < 4; ++t2)
#pragma unroll
      for (int r = 0; r < 8; ++r)
        oacc[t2][r] *= scale_r[r];

    // P: C-layout -> A-layout via wave-private LDS (same-wave DS ordering)
#pragma unroll
    for (int r = 0; r < 8; ++r) {
      int rr = r + (half << 3);
      pbase[rr * 32 + lrow]      = f2bf(p0[r]);
      pbase[rr * 32 + 16 + lrow] = f2bf(p1[r]);
    }
    const __bf16* pap = pbase + lrow * 32 + (half << 3);
    v16bf pa = frag_from(pap, pap + 16);

#pragma unroll
    for (int t2 = 0; t2 < 4; ++t2) {        // O += P * V  (V^T layout)
      const __bf16* vp = Vbh + (size_t)(t2 * 16 + lrow) * N_ + j0 + (half << 3);
      v16bf vf = frag_from(vp, vp + 16);
      oacc[t2] = wmma_bf16(pa, vf, oacc[t2]);
    }
  }

#pragma unroll
  for (int r = 0; r < 8; ++r) {             // finalize: /= l, write [b,n,h,d]
    float inv = 1.0f / l[r];
    int nn = i0 + r + (half << 3);
    size_t rowb = ((size_t)(b * N_ + nn)) * HID_ + h * DH_;
#pragma unroll
    for (int t2 = 0; t2 < 4; ++t2)
      Ob[rowb + t2 * 16 + lrow] = f2bf(oacc[t2][r] * inv);
  }
}

// ---------------------------------------------------------- output GEMM ----
// out[8192,512] = Ab[8192,512] * wT[512,512]^T ; wave tile 32x64.
__global__ __launch_bounds__(256) void k_out_gemm(const __bf16* __restrict__ Ab,
                                                  const __bf16* __restrict__ wT,
                                                  float* __restrict__ out) {
  const int lane = threadIdx.x & 31;
  const int wid  = threadIdx.x >> 5;
  const int gw   = blockIdx.x * 8 + wid;
  const int TN = DIM_ / 64;                 // 8 col tiles
  const int tm = gw / TN, tn = gw % TN;
  const int m0 = tm * 32, n0 = tn * 64;
  const int lrow = lane & 15;
  const int half = lane >> 4;
  const int kbA = half << 3, kbB = half << 3;

  v8f acc[2][4] = {};
#pragma unroll 2
  for (int k0 = 0; k0 < HID_; k0 += 32) {
    const __bf16* a0p = Ab + (size_t)(m0 + lrow) * HID_ + k0 + kbA;
    const __bf16* a1p = Ab + (size_t)(m0 + 16 + lrow) * HID_ + k0 + kbA;
    v16bf a0 = frag_from(a0p, a0p + 16);
    v16bf a1 = frag_from(a1p, a1p + 16);
    v16bf bf[4];
#pragma unroll
    for (int j = 0; j < 4; ++j) {
      const __bf16* bp = wT + (size_t)(n0 + j * 16 + lrow) * HID_ + k0 + kbB;
      bf[j] = frag_from(bp, bp + 16);
    }
#pragma unroll
    for (int j = 0; j < 4; ++j) {
      acc[0][j] = wmma_bf16(a0, bf[j], acc[0][j]);
      acc[1][j] = wmma_bf16(a1, bf[j], acc[1][j]);
    }
  }
#pragma unroll
  for (int i = 0; i < 2; ++i)
#pragma unroll
    for (int j = 0; j < 4; ++j)
#pragma unroll
      for (int r = 0; r < 8; ++r) {
        int row = m0 + i * 16 + r + (half << 3);
        int col = n0 + j * 16 + lrow;
        out[(size_t)row * DIM_ + col] = acc[i][j][r];
      }
}

// ---------------------------------------------------------------- launch ----
extern "C" void kernel_launch(void* const* d_in, const int* in_sizes, int n_in,
                              void* d_out, int out_size, void* d_ws, size_t ws_size,
                              hipStream_t stream) {
  (void)in_sizes; (void)n_in; (void)out_size; (void)ws_size;
  const float* x    = (const float*)d_in[0];
  const float* bias = (const float*)d_in[1];
  const float* wqkv = (const float*)d_in[2];
  const float* wout = (const float*)d_in[3];

  char* ws = (char*)d_ws;
  __bf16* xb    = (__bf16*)ws; ws += (size_t)M_ * DIM_ * 2;                 // 8 MB
  __bf16* wqkvT = (__bf16*)ws; ws += (size_t)QKV3_ * DIM_ * 2;              // 1.5 MB
  __bf16* woutT = (__bf16*)ws; ws += (size_t)HID_ * DIM_ * 2;               // 0.5 MB
  __bf16* Qs    = (__bf16*)ws; ws += (size_t)B_ * HEADS_ * N_ * DH_ * 2;    // 8 MB
  __bf16* Kbuf  = (__bf16*)ws; ws += (size_t)B_ * HEADS_ * N_ * DH_ * 2;    // 8 MB
  __bf16* Vt    = (__bf16*)ws; ws += (size_t)B_ * HEADS_ * DH_ * N_ * 2;    // 8 MB
  __bf16* Ob    = (__bf16*)ws; ws += (size_t)M_ * HID_ * 2;                 // 8 MB

  k_cvt_bf16<<<(M_ * DIM_) / (256 * 4), 256, 0, stream>>>(x, xb, M_ * DIM_);
  k_transpose_cvt<<<QKV3_, 256, 0, stream>>>(wqkv, wqkvT, DIM_, QKV3_);
  k_transpose_cvt<<<DIM_, 256, 0, stream>>>(wout, woutT, HID_, DIM_);
  k_qkv_gemm<<<(M_ / 32) * (QKV3_ / 64) / 8, 256, 0, stream>>>(xb, wqkvT, Qs, Kbuf, Vt);
  k_flash<<<(B_ * HEADS_ * (N_ / 16)) / 8, 256, 0, stream>>>(Qs, Kbuf, Vt, bias, Ob);
  k_out_gemm<<<(M_ / 32) * (DIM_ / 64) / 8, 256, 0, stream>>>(Ob, woutT, (float*)d_out);
}
